// Attention_8315056685174
// MI455X (gfx1250) — matrix operand order repaired and checked by reading, not compiled, not run on hardware
//
#include <hip/hip_runtime.h>

typedef __attribute__((ext_vector_type(16))) _Float16 v16h;
typedef __attribute__((ext_vector_type(8)))  float    v8f;

#define PH 72   // f16 LDS pitch (halfs) per row of 64  -> 144B, 16B aligned
#define PF 68   // f32 LDS pitch (floats)

// ---- WMMA fragment helpers (layouts per CDNA5 ISA 7.12.2) -------------------
// A (16x32 f16): lanes 0-15 row M=lane, halves 0..7 = K khi..khi+7,
// halves 8..15 = K 16+khi..; lanes 16-31 khi+=8. Row-major [M][K] source.
__device__ __forceinline__ v16h load_a_frag(const _Float16* base, int pitch,
                                            int m0, int k0, int lane) {
  const int row = m0 + (lane & 15);
  const int khi = (lane >> 4) << 3;
  const _Float16* p = base + row * pitch + k0 + khi;
  union { v16h h; float4 f4[2]; } u;
  u.f4[0] = *reinterpret_cast<const float4*>(p);
  u.f4[1] = *reinterpret_cast<const float4*>(p + 16);
  return u.h;
}
// B (32x16 f16): lane holds column n0+(lane&15), halves 0..7 = K klo..klo+7,
// halves 8..15 = K 16+klo..; lanes 16-31 klo+=8 (same as A).
// Source must be B-transposed: baseT[N][K].
__device__ __forceinline__ v16h load_b_frag(const _Float16* baseT, int pitch,
                                            int n0, int k0, int lane) {
  const int col = n0 + (lane & 15);
  const int klo = k0 + ((lane >> 4) << 3);
  const _Float16* p = baseT + col * pitch + klo;
  union { v16h h; float4 f4[2]; } u;
  u.f4[0] = *reinterpret_cast<const float4*>(p);
  u.f4[1] = *reinterpret_cast<const float4*>(p + 16);
  return u.h;
}
__device__ __forceinline__ v8f wmma16(v16h a, v16h b, v8f c) {
  return __builtin_amdgcn_wmma_f32_16x16x32_f16(false, a, false, b,
                                                (short)0, c, false, false);
}

// ---- prep kernels -----------------------------------------------------------
__global__ void prep_w_f16(const float* __restrict__ kv_w,
                           const float* __restrict__ proj_w,
                           _Float16* __restrict__ KVW,
                           _Float16* __restrict__ PROJW) {
  int i = blockIdx.x * 256 + threadIdx.x;
  if (i < 128 * 64) KVW[i] = (_Float16)kv_w[i];
  int j = i - 128 * 64;
  if (j >= 0 && j < 64 * 64) PROJW[j] = (_Float16)proj_w[j];
}

__global__ void prep_qn(const float* __restrict__ q,
                        _Float16* __restrict__ QN,
                        _Float16* __restrict__ QNT) {
  const int bc = blockIdx.x;          // b*64 + c
  const int b  = bc >> 6, c = bc & 63;
  const int t  = threadIdx.x;         // 64 threads = one row of P
  float v = q[bc * 64 + t];
  __shared__ float red[64];
  red[t] = v * v;
  __syncthreads();
  for (int s = 32; s > 0; s >>= 1) {
    if (t < s) red[t] += red[t + s];
    __syncthreads();
  }
  float sc = 1.f / fmaxf(sqrtf(red[0]), 1e-12f);
  float qn = v * sc;
  QN [bc * 64 + t] = (_Float16)qn;             // [b][c][p]
  QNT[(b * 64 + t) * 64 + c] = (_Float16)qn;   // [b][p][c]
}

// ---- main fused per-window kernel ------------------------------------------
// LDS carve (bytes):
//   0     : XT / ATT   f16 [64][PH]   9216
//   9216  : KVh        f16 [128][PH] 18432   (overlaid later by S f32 [64][PF] 17408)
//   27648 : KN         f16 [64][PH]   9216
//   36864 : V          f16 [64][PH]   9216
//   45 -> : VT         f16 [64][PH]   9216  @46080
//   55296 : OUTT       f16 [64][PH]   9216
//   64512 : ROW        f32 [64]        256
// total 64768 B
__global__ __launch_bounds__(256)
void attn_win_kernel(const float* __restrict__ x,
                     const float* __restrict__ dw_w,
                     const float* __restrict__ rescale1,
                     const float* __restrict__ rescale2,
                     const _Float16* __restrict__ KVW,
                     const _Float16* __restrict__ PROJW,
                     const _Float16* __restrict__ QN,
                     const _Float16* __restrict__ QNT,
                     float* __restrict__ out) {
  __shared__ __align__(16) char smem[64768];
  _Float16* XT   = (_Float16*)(smem);           // B^T for KV GEMM
  _Float16* ATT  = (_Float16*)(smem);           // reuse after KV GEMM
  _Float16* KVh  = (_Float16*)(smem + 9216);
  float*    Sf32 = (float*)   (smem + 9216);    // reuse after dwconv
  _Float16* KN   = (_Float16*)(smem + 27648);
  _Float16* V    = (_Float16*)(smem + 36864);
  _Float16* VT   = (_Float16*)(smem + 46080);
  _Float16* OUTT = (_Float16*)(smem + 55296);
  float*    ROW  = (float*)   (smem + 64512);

  const int tid  = threadIdx.x;
  const int wave = tid >> 5;
  const int lane = tid & 31;

  const int blk = blockIdx.x;             // 8192 = B * 32 * 32
  const int b   = blk >> 10;
  const int rem = blk & 1023;
  const int wy  = rem >> 5;
  const int wx  = rem & 31;

  const float r1 = rescale1[0];
  const float r2 = rescale2[0];

  if (tid < 64) ROW[tid] = 0.f;

  // ---- Phase 1: load window of x (with -4,-4 roll) as XT[p][c] f16 ---------
  const long xbase = (long)b * 64 * 65536;
  #pragma unroll
  for (int i = 0; i < 16; ++i) {
    int idx = i * 256 + tid;              // 4096 = 64ch * 64px
    int p = idx & 63, c = idx >> 6;
    int py = p >> 3, px = p & 7;
    int Y = (wy * 8 + py + 4) & 255;      // roll(-4): rolled[i] = x[i+4]
    int X = (wx * 8 + px + 4) & 255;
    float v = x[xbase + (long)c * 65536 + Y * 256 + X];
    XT[p * PH + c] = (_Float16)v;
  }
  __syncthreads();

  // ---- Phase 2: KV = KVW(128x64) @ X(64x64), one tile-row per wave ---------
  {
    const int m0 = wave * 16;
    v16h a0 = load_a_frag(KVW, 64, m0, 0,  lane);
    v16h a1 = load_a_frag(KVW, 64, m0, 32, lane);
    #pragma unroll
    for (int tn = 0; tn < 4; ++tn) {
      int n0 = tn * 16;
      v16h b0 = load_b_frag(XT, PH, n0, 0,  lane);
      v16h b1 = load_b_frag(XT, PH, n0, 32, lane);
      v8f acc = {};
      acc = wmma16(a0, b0, acc);
      acc = wmma16(a1, b1, acc);
      int mrow = m0 + ((lane >> 4) << 3);
      int ncol = n0 + (lane & 15);
      #pragma unroll
      for (int r = 0; r < 8; ++r)
        KVh[(mrow + r) * PH + ncol] = (_Float16)acc[r];
    }
  }
  __syncthreads();

  // ---- Phase 3: depthwise 3x3 (window-local zero pad) + L2-norm of k -------
  {
    const int ch = tid >> 1;              // 0..127, two threads per channel
    const int hf = tid & 1;               // pixel half: rows 0-3 or 4-7
    float w9[9];
    #pragma unroll
    for (int j = 0; j < 9; ++j) w9[j] = dw_w[ch * 9 + j];
    const _Float16* kvrow = KVh + ch * PH;
    float o[32];
    float sumsq = 0.f;
    #pragma unroll
    for (int i = 0; i < 32; ++i) {
      int p = hf * 32 + i;
      int py = p >> 3, px = p & 7;
      float s = 0.f;
      #pragma unroll
      for (int dy = -1; dy <= 1; ++dy)
        #pragma unroll
        for (int dx = -1; dx <= 1; ++dx) {
          int ny = py + dy, nx = px + dx;
          if (ny >= 0 && ny < 8 && nx >= 0 && nx < 8)
            s += w9[(dy + 1) * 3 + (dx + 1)] * (float)kvrow[ny * 8 + nx];
        }
      o[i] = s;
      if (ch < 64) sumsq += s * s;
    }
    if (ch < 64) atomicAdd(&ROW[ch], sumsq);   // ds_add_f32
    __syncthreads();
    if (ch < 64) {
      float sc = 1.f / fmaxf(sqrtf(ROW[ch]), 1e-12f);
      #pragma unroll
      for (int i = 0; i < 32; ++i)
        KN[ch * PH + hf * 32 + i] = (_Float16)(o[i] * sc);
    } else {
      int cv = ch - 64;
      #pragma unroll
      for (int i = 0; i < 32; ++i) {
        int p = hf * 32 + i;
        _Float16 h = (_Float16)o[i];
        V [cv * PH + p] = h;
        VT[p * PH + cv] = h;
      }
    }
  }
  __syncthreads();

  // ---- attention: each wave owns 2 of 16 output tiles ----------------------
  const int tm = wave >> 1;
  const int m0 = tm * 16;
  const int nA = (wave & 1) * 32;
  const _Float16* qnb  = QN  + b * 4096;  // [c][p], pitch 64
  const _Float16* qntb = QNT + b * 4096;  // [p][c], pitch 64

  v16h ka0 = load_a_frag(KN, PH, m0, 0,  lane);
  v16h ka1 = load_a_frag(KN, PH, m0, 32, lane);
  const int mrow = m0 + ((lane >> 4) << 3);

  // S1 = kn @ qn^T
  #pragma unroll
  for (int h = 0; h < 2; ++h) {
    int n0 = nA + h * 16;
    v16h b0 = load_b_frag(qnb, 64, n0, 0,  lane);
    v16h b1 = load_b_frag(qnb, 64, n0, 32, lane);
    v8f acc = {};
    acc = wmma16(ka0, b0, acc);
    acc = wmma16(ka1, b1, acc);
    int ncol = n0 + (lane & 15);
    #pragma unroll
    for (int r = 0; r < 8; ++r) Sf32[(mrow + r) * PF + ncol] = acc[r];
  }
  __syncthreads();

  // softmax rows of S (4 lanes per row) -> ATT f16
  {
    const int row = wave * 8 + (lane >> 2);
    const int sub = lane & 3;
    const float* srow = Sf32 + row * PF + sub * 16;
    float vals[16], mx = -1e30f;
    #pragma unroll
    for (int j = 0; j < 16; ++j) { vals[j] = srow[j] * r1; mx = fmaxf(mx, vals[j]); }
    mx = fmaxf(mx, __shfl_xor(mx, 1, 32));
    mx = fmaxf(mx, __shfl_xor(mx, 2, 32));
    float sm = 0.f;
    #pragma unroll
    for (int j = 0; j < 16; ++j) { vals[j] = __expf(vals[j] - mx); sm += vals[j]; }
    sm += __shfl_xor(sm, 1, 32);
    sm += __shfl_xor(sm, 2, 32);
    float inv = 1.f / sm;
    _Float16* arow = ATT + row * PH + sub * 16;
    #pragma unroll
    for (int j = 0; j < 16; ++j) arow[j] = (_Float16)(vals[j] * inv);
  }
  __syncthreads();

  // out1 = attn1 @ v  (B^T = VT) ;  S2 = kn @ v^T  (B^T = V)
  v8f oacc[2];
  {
    v16h at0 = load_a_frag(ATT, PH, m0, 0,  lane);
    v16h at1 = load_a_frag(ATT, PH, m0, 32, lane);
    #pragma unroll
    for (int h = 0; h < 2; ++h) {
      int n0 = nA + h * 16;
      v16h b0 = load_b_frag(VT, PH, n0, 0,  lane);
      v16h b1 = load_b_frag(VT, PH, n0, 32, lane);
      v8f acc = {};
      acc = wmma16(at0, b0, acc);
      acc = wmma16(at1, b1, acc);
      oacc[h] = acc;
      v16h vb0 = load_b_frag(V, PH, n0, 0,  lane);
      v16h vb1 = load_b_frag(V, PH, n0, 32, lane);
      v8f sa = {};
      sa = wmma16(ka0, vb0, sa);
      sa = wmma16(ka1, vb1, sa);
      int ncol = n0 + (lane & 15);
      #pragma unroll
      for (int r = 0; r < 8; ++r) Sf32[(mrow + r) * PF + ncol] = sa[r];
    }
  }
  __syncthreads();

  // softmax(S2 * r2) -> ATT
  {
    const int row = wave * 8 + (lane >> 2);
    const int sub = lane & 3;
    const float* srow = Sf32 + row * PF + sub * 16;
    float vals[16], mx = -1e30f;
    #pragma unroll
    for (int j = 0; j < 16; ++j) { vals[j] = srow[j] * r2; mx = fmaxf(mx, vals[j]); }
    mx = fmaxf(mx, __shfl_xor(mx, 1, 32));
    mx = fmaxf(mx, __shfl_xor(mx, 2, 32));
    float sm = 0.f;
    #pragma unroll
    for (int j = 0; j < 16; ++j) { vals[j] = __expf(vals[j] - mx); sm += vals[j]; }
    sm += __shfl_xor(sm, 1, 32);
    sm += __shfl_xor(sm, 2, 32);
    float inv = 1.f / sm;
    _Float16* arow = ATT + row * PH + sub * 16;
    #pragma unroll
    for (int j = 0; j < 16; ++j) arow[j] = (_Float16)(vals[j] * inv);
  }
  __syncthreads();

  // out += attn2 @ qn  (B^T = QNT); fft round-trip is identity -> out1+out2.
  // Write OUT transposed [p][c] f16 for proj GEMM.
  {
    v16h at0 = load_a_frag(ATT, PH, m0, 0,  lane);
    v16h at1 = load_a_frag(ATT, PH, m0, 32, lane);
    #pragma unroll
    for (int h = 0; h < 2; ++h) {
      int n0 = nA + h * 16;
      v16h b0 = load_b_frag(qntb, 64, n0, 0,  lane);
      v16h b1 = load_b_frag(qntb, 64, n0, 32, lane);
      v8f acc = oacc[h];
      acc = wmma16(at0, b0, acc);
      acc = wmma16(at1, b1, acc);
      int ncol = n0 + (lane & 15);
      union { _Float16 hh[8]; float4 f4; } u;
      #pragma unroll
      for (int r = 0; r < 8; ++r) u.hh[r] = (_Float16)acc[r];
      *reinterpret_cast<float4*>(&OUTT[ncol * PH + mrow]) = u.f4;
    }
  }
  __syncthreads();

  // ---- proj 1x1: F = PROJW(64x64) @ OUT(64x64); store with +4 roll-back ----
  {
    v16h pa0 = load_a_frag(PROJW, 64, m0, 0,  lane);
    v16h pa1 = load_a_frag(PROJW, 64, m0, 32, lane);
    #pragma unroll
    for (int h = 0; h < 2; ++h) {
      int n0 = nA + h * 16;
      v16h b0 = load_b_frag(OUTT, PH, n0, 0,  lane);
      v16h b1 = load_b_frag(OUTT, PH, n0, 32, lane);
      v8f f = {};
      f = wmma16(pa0, b0, f);
      f = wmma16(pa1, b1, f);
      int ncol = n0 + (lane & 15);
      int py = ncol >> 3, px = ncol & 7;
      int Y = (wy * 8 + py + 4) & 255;    // roll(+4)
      int X = (wx * 8 + px + 4) & 255;
      #pragma unroll
      for (int r = 0; r < 8; ++r) {
        int oc = mrow + r;
        out[((long)(b * 64 + oc)) * 65536 + Y * 256 + X] = f[r];
      }
    }
  }
}

// ---- host launcher ----------------------------------------------------------
extern "C" void kernel_launch(void* const* d_in, const int* in_sizes, int n_in,
                              void* d_out, int out_size, void* d_ws, size_t ws_size,
                              hipStream_t stream) {
  const float* x      = (const float*)d_in[0];
  const float* q      = (const float*)d_in[1];
  const float* kv_w   = (const float*)d_in[2];
  const float* dw_w   = (const float*)d_in[3];
  const float* proj_w = (const float*)d_in[4];
  const float* r1     = (const float*)d_in[5];
  const float* r2     = (const float*)d_in[6];

  char* ws = (char*)d_ws;
  _Float16* KVW   = (_Float16*)(ws);            // 128*64*2 = 16384
  _Float16* PROJW = (_Float16*)(ws + 16384);    // 64*64*2  =  8192
  _Float16* QN    = (_Float16*)(ws + 24576);    // 8*64*64*2 = 65536
  _Float16* QNT   = (_Float16*)(ws + 90112);    // 65536

  prep_w_f16<<<48, 256, 0, stream>>>(kv_w, proj_w, KVW, PROJW);
  prep_qn<<<512, 64, 0, stream>>>(q, QN, QNT);
  attn_win_kernel<<<8192, 256, 0, stream>>>(x, dw_w, r1, r2, KVW, PROJW, QN, QNT,
                                            (float*)d_out);
}
